// TransformerBlock_798863917769
// MI455X (gfx1250) — matrix operand order repaired and checked by reading, not compiled, not run on hardware
//
#include <hip/hip_runtime.h>

// ---------------- problem constants ----------------
#define B_   2
#define L_   2048
#define E_   1024
#define NH_  16
#define DH_  64
#define HID_ 4096
#define M_   (B_ * L_)        // 4096 rows of activations

#define DEV static __device__ __forceinline__

typedef __bf16 bf16;
typedef __attribute__((ext_vector_type(8)))  bf16  bf16x8;
typedef __attribute__((ext_vector_type(16))) bf16  v16bf;
typedef __attribute__((ext_vector_type(8)))  float v8f;
typedef unsigned int u32;
typedef __attribute__((ext_vector_type(4))) u32 u32x4;
typedef __attribute__((ext_vector_type(4))) int i32x4;
typedef __attribute__((ext_vector_type(8))) int i32x8;

#if defined(__has_builtin)
#if __has_builtin(__builtin_amdgcn_tensor_load_to_lds)
#define HAVE_TDM 1
#else
#define HAVE_TDM 0
#endif
#else
#define HAVE_TDM 0
#endif

// ---------------- helpers ----------------
DEV bf16 f2bf(float f) {
    unsigned u = __builtin_bit_cast(unsigned, f);
    unsigned r = u + 0x7FFFu + ((u >> 16) & 1u);   // round-to-nearest-even
    unsigned short s = (unsigned short)(r >> 16);
    return __builtin_bit_cast(bf16, s);
}

DEV v8f zero8() { v8f z; for (int i = 0; i < 8; ++i) z[i] = 0.0f; return z; }

// A-fragment (16x32 bf16, row-major source, row = lane%16):
// lanes 0-15 hold K=[0..15]; lanes 16-31 hold K=[16..31]
DEV v16bf load_afrag(const bf16* rowptr, int hl) {
    bf16x8 lo = *reinterpret_cast<const bf16x8*>(rowptr + hl * 16);
    bf16x8 hi = *reinterpret_cast<const bf16x8*>(rowptr + hl * 16 + 8);
    return __builtin_shufflevector(lo, hi, 0,1,2,3,4,5,6,7,8,9,10,11,12,13,14,15);
}

// B-fragment (32x16 bf16): lane = column, elements = K (0-15 lanes: K0..15,
// 16-31 lanes: K16..31). colptr points at column's K-contiguous data (N-major).
DEV v16bf load_bfrag(const bf16* colptr, int hl) {
    return *reinterpret_cast<const v16bf*>(colptr + hl * 16);
}

DEV v8f wmma_bf16(v16bf a, v16bf b, v8f c) {
    // (neg_a, A, neg_b, B, c_mod, C, reuse_a, reuse_b)
    return __builtin_amdgcn_wmma_f32_16x16x32_bf16(false, a, false, b,
                                                   (short)0, c, false, false);
}

DEV void wait_tensor0() {
#if defined(__has_builtin) && __has_builtin(__builtin_amdgcn_s_wait_tensorcnt)
    __builtin_amdgcn_s_wait_tensorcnt(0);
#else
    asm volatile("s_wait_tensorcnt 0x0" ::: "memory");
#endif
}

#if HAVE_TDM
// Program a 2D TDM descriptor and issue TENSOR_LOAD_TO_LDS:
// tile = tileRows x 32 bf16 elements, row stride = rowStrideElems, packed into LDS.
DEV void tdm_load_2d(const bf16* gsrc, const bf16* ldsDst,
                     u32 tileRows, u32 rowStrideElems) {
    unsigned long long ga = (unsigned long long)(__SIZE_TYPE__)(const void*)gsrc;
    u32 lds = (u32)(__SIZE_TYPE__)(const void*)ldsDst;   // low 32 bits = LDS offset
    u32x4 g0;
    g0[0] = 1u;                                           // count = 1 descriptor
    g0[1] = lds;                                          // lds_addr (bytes)
    g0[2] = (u32)ga;                                      // global_addr[31:0]
    g0[3] = ((u32)(ga >> 32) & 0x01FFFFFFu) | (2u << 30); // global_addr[56:32] | type=2
    i32x8 g1;
    g1[0] = 0x00010000;                                   // data_size=1 (2 bytes)
    g1[1] = (int)((rowStrideElems & 0xFFFFu) << 16);      // tensor_dim0[15:0]
    g1[2] = (int)((rowStrideElems >> 16) & 0xFFFFu);      // tensor_dim0[31:16], dim1 lo=0
    g1[3] = (int)(16u | (32u << 16));                     // tensor_dim1 hi (=1M rows), tile_dim0=32
    g1[4] = (int)tileRows;                                // tile_dim1, tile_dim2=0
    g1[5] = (int)rowStrideElems;                          // tensor_dim0_stride[31:0]
    g1[6] = 0;                                            // stride hi / dim1_stride
    g1[7] = 0;
    i32x4 z4; z4[0] = 0; z4[1] = 0; z4[2] = 0; z4[3] = 0;
#if defined(__clang_major__) && (__clang_major__ >= 23)
    i32x8 z8; for (int i = 0; i < 8; ++i) z8[i] = 0;
    __builtin_amdgcn_tensor_load_to_lds(g0, g1, z4, z4, z8, 0);
#else
    __builtin_amdgcn_tensor_load_to_lds(g0, g1, z4, z4, 0);
#endif
}
#endif

// ---------------- weight convert + transpose: W[K][N] f32 -> Wt[N][K] bf16 ----------------
__global__ __launch_bounds__(256) void wtrans_kernel(const float* __restrict__ W,
                                                     bf16* __restrict__ Wt,
                                                     int K, int N) {
    long idx = (long)blockIdx.x * 256 + threadIdx.x;
    if (idx >= (long)K * N) return;
    int n = (int)(idx / K);
    int k = (int)(idx % K);
    Wt[idx] = f2bf(W[(size_t)k * N + n]);
}

// ---------------- layernorm: f32 in -> bf16 out, one block per row ----------------
__global__ __launch_bounds__(256) void ln_kernel(const float* __restrict__ x,
                                                 const float* __restrict__ g,
                                                 const float* __restrict__ be,
                                                 bf16* __restrict__ out) {
    __shared__ float s1[256], s2[256];
    const int row = blockIdx.x;
    const float* xr = x + (size_t)row * E_;
    float s = 0.f, q = 0.f;
    for (int i = threadIdx.x; i < E_; i += 256) { float v = xr[i]; s += v; q += v * v; }
    s1[threadIdx.x] = s; s2[threadIdx.x] = q;
    __syncthreads();
    for (int off = 128; off > 0; off >>= 1) {
        if ((int)threadIdx.x < off) {
            s1[threadIdx.x] += s1[threadIdx.x + off];
            s2[threadIdx.x] += s2[threadIdx.x + off];
        }
        __syncthreads();
    }
    const float mu   = s1[0] * (1.0f / E_);
    const float var  = s2[0] * (1.0f / E_) - mu * mu;
    const float rstd = rsqrtf(var + 1e-5f);
    for (int i = threadIdx.x; i < E_; i += 256)
        out[(size_t)row * E_ + i] = f2bf((xr[i] - mu) * rstd * g[i] + be[i]);
}

// ---------------- generic WMMA GEMM, TDM-staged tiles ----------------
// C[M,N] = A[M,K](bf16) * Bt[N,K](bf16) (+bias) (+resid)
// 128 threads = 4 waves; block tile 64(M) x 128(N); wave tile 32x64 (2x4 frags).
// TDM stages A(64x32) and B(128x32) K-slabs into LDS; waves eat frags via DS.
// EPI: 0 = bf16 out (no bias) | 1 = bf16 out + bias | 2 = f32 out + bias + resid
template <int EPI>
__global__ __launch_bounds__(128) void gemm_wmma_kernel(const bf16* __restrict__ A,
                                                        const bf16* __restrict__ Bt,
                                                        const float* __restrict__ bias,
                                                        const float* __restrict__ resid,
                                                        bf16* __restrict__ outb,
                                                        float* __restrict__ outf,
                                                        int M, int N, int K) {
    const int lane = threadIdx.x & 31;
    const int wave = threadIdx.x >> 5;
    const int hl   = lane >> 4;
    const int l16  = lane & 15;
    const int mBase = blockIdx.y * 64;
    const int nBase = blockIdx.x * 128;
    const int wm = (wave >> 1) * 32;     // wave offset in block (M)
    const int wn = (wave & 1) * 64;      // wave offset in block (N)
    (void)M;

    v8f acc[2][4];
    for (int mt = 0; mt < 2; ++mt)
        for (int nt = 0; nt < 4; ++nt) acc[mt][nt] = zero8();

#if HAVE_TDM
    __shared__ __align__(128) bf16 ldsA[64 * 32];
    __shared__ __align__(128) bf16 ldsB[128 * 32];
    for (int k0 = 0; k0 < K; k0 += 32) {
        __syncthreads();   // previous slab fully consumed before DMA overwrites
        // each wave DMAs its own slice; TENSORcnt is per-wave
        tdm_load_2d(A  + (size_t)(mBase + wave * 16) * K + k0, ldsA + wave * 16 * 32,
                    16, (u32)K);
        tdm_load_2d(Bt + (size_t)(nBase + wave * 32) * K + k0, ldsB + wave * 32 * 32,
                    32, (u32)K);
        wait_tensor0();
        __syncthreads();

        v16bf af[2], bfr[4];
        for (int mt = 0; mt < 2; ++mt)
            af[mt] = load_afrag(ldsA + (wm + mt * 16 + l16) * 32, hl);
        for (int nt = 0; nt < 4; ++nt)
            bfr[nt] = load_bfrag(ldsB + (wn + nt * 16 + l16) * 32, hl);
        for (int mt = 0; mt < 2; ++mt)
            for (int nt = 0; nt < 4; ++nt)
                acc[mt][nt] = wmma_bf16(af[mt], bfr[nt], acc[mt][nt]);
    }
#else
    const bf16* ap[2];
    const bf16* bp[4];
    for (int mt = 0; mt < 2; ++mt) ap[mt] = A  + (size_t)(mBase + wm + mt * 16 + l16) * K;
    for (int nt = 0; nt < 4; ++nt) bp[nt] = Bt + (size_t)(nBase + wn + nt * 16 + l16) * K;
    for (int k0 = 0; k0 < K; k0 += 32) {
        __builtin_prefetch(ap[0] + k0 + 128, 0, 1);
        __builtin_prefetch(bp[0] + k0 + 128, 0, 1);
        v16bf af[2], bfr[4];
        for (int mt = 0; mt < 2; ++mt) af[mt]  = load_afrag(ap[mt] + k0, hl);
        for (int nt = 0; nt < 4; ++nt) bfr[nt] = load_bfrag(bp[nt] + k0, hl);
        for (int mt = 0; mt < 2; ++mt)
            for (int nt = 0; nt < 4; ++nt)
                acc[mt][nt] = wmma_bf16(af[mt], bfr[nt], acc[mt][nt]);
    }
#endif

    for (int mt = 0; mt < 2; ++mt)
        for (int nt = 0; nt < 4; ++nt)
            for (int r = 0; r < 8; ++r) {
                const int row = mBase + wm + mt * 16 + hl * 8 + r;
                const int col = nBase + wn + nt * 16 + l16;
                float v = acc[mt][nt][r];
                if (EPI >= 1) v += bias[col];
                if (EPI == 2) {
                    v += resid[(size_t)row * N + col];
                    outf[(size_t)row * N + col] = v;
                } else {
                    outb[(size_t)row * N + col] = f2bf(v);
                }
            }
}

// ---------------- V transpose: qkv -> vt[(b*NH+h)*DH + d][L] ----------------
__global__ __launch_bounds__(256) void build_vt_kernel(const bf16* __restrict__ qkv,
                                                       bf16* __restrict__ vt) {
    size_t idx = (size_t)blockIdx.x * 256 + threadIdx.x; // B*NH*DH*L = 2^22
    int kv = (int)(idx & (L_ - 1));
    int d  = (int)((idx >> 11) & (DH_ - 1));
    int h  = (int)((idx >> 17) & (NH_ - 1));
    int b  = (int)(idx >> 21);
    vt[idx] = qkv[((size_t)(b * L_ + kv)) * (NH_ * 3 * DH_) + h * (3 * DH_) + 2 * DH_ + d];
}

// ---------------- flash-style causal attention, one wave per 16-query tile ----------------
__global__ __launch_bounds__(128) void attn_kernel(const bf16* __restrict__ qkv,
                                                   const bf16* __restrict__ vt,
                                                   bf16* __restrict__ attnout) {
    __shared__ __align__(64) bf16 Pbuf[4][16 * 32];
    const int lane = threadIdx.x & 31;
    const int wave = threadIdx.x >> 5;
    const int hl   = lane >> 4;
    const int l16  = lane & 15;

    const int job = blockIdx.x * 4 + wave;       // 4096 jobs = B*NH*(L/16)
    const int qt  = job & 127;
    const int h   = (job >> 7) & (NH_ - 1);
    const int b   = job >> 11;
    const int q0  = qt * 16;

    const int qkvLd = NH_ * 3 * DH_;             // 3072
    const bf16* qrow = qkv + (size_t)(b * L_ + q0 + l16) * qkvLd + h * (3 * DH_);
    const v16bf qa0 = load_afrag(qrow, hl);      // d = 0..31
    const v16bf qa1 = load_afrag(qrow + 32, hl); // d = 32..63

    float m_r[8], l_r[8];
    v8f o[4];
    for (int r = 0; r < 8; ++r) { m_r[r] = -3.0e38f; l_r[r] = 0.0f; }
    for (int c = 0; c < 4; ++c) o[c] = zero8();

    bf16* Pw = &Pbuf[wave][0];

    for (int kv0 = 0; kv0 < q0 + 16; kv0 += 32) {
        float sreg[2][8];
        float tmax[8];
        for (int r = 0; r < 8; ++r) tmax[r] = -3.0e38f;

        for (int t = 0; t < 2; ++t) {
            const int key  = kv0 + t * 16 + l16;
            const int keyc = key < (L_ - 1) ? key : (L_ - 1);   // clamp for addr safety
            const bf16* kb = qkv + (size_t)(b * L_ + keyc) * qkvLd + h * (3 * DH_) + DH_;
            v16bf kb0 = *reinterpret_cast<const v16bf*>(kb + hl * 16);       // d 0..31
            v16bf kb1 = *reinterpret_cast<const v16bf*>(kb + 32 + hl * 16);  // d 32..63
            v8f s = zero8();
            s = wmma_bf16(qa0, kb0, s);
            s = wmma_bf16(qa1, kb1, s);
            for (int r = 0; r < 8; ++r) {
                const int row = q0 + hl * 8 + r;
                float v = s[r] * 0.125f;                 // 1/sqrt(64)
                if (key > row) v = -3.0e38f;             // causal mask
                sreg[t][r] = v;
                tmax[r] = fmaxf(tmax[r], v);
            }
        }

        // online softmax on the C-fragment layout (row = hl*8 + r, 16 lanes/half)
        for (int r = 0; r < 8; ++r) {
            float v = tmax[r];
            v = fmaxf(v, __shfl_xor(v, 1));
            v = fmaxf(v, __shfl_xor(v, 2));
            v = fmaxf(v, __shfl_xor(v, 4));
            v = fmaxf(v, __shfl_xor(v, 8));
            const float mn    = fmaxf(m_r[r], v);
            const float alpha = __expf(m_r[r] - mn);
            m_r[r] = mn;
            const float p0 = __expf(sreg[0][r] - mn);
            const float p1 = __expf(sreg[1][r] - mn);
            float rs = p0 + p1;
            rs += __shfl_xor(rs, 1);
            rs += __shfl_xor(rs, 2);
            rs += __shfl_xor(rs, 4);
            rs += __shfl_xor(rs, 8);
            l_r[r] = l_r[r] * alpha + rs;
            for (int c = 0; c < 4; ++c) o[c][r] *= alpha;
            Pw[(hl * 8 + r) * 32 + l16]      = f2bf(p0);
            Pw[(hl * 8 + r) * 32 + 16 + l16] = f2bf(p1);
        }
        __builtin_amdgcn_wave_barrier();     // DS in-order per wave; fence compiler

        // re-read P as an A-fragment (16x32)
        const bf16* prow = Pw + l16 * 32;
        bf16x8 plo = *reinterpret_cast<const bf16x8*>(prow + hl * 16);
        bf16x8 phi = *reinterpret_cast<const bf16x8*>(prow + hl * 16 + 8);
        v16bf pa = __builtin_shufflevector(plo, phi,
                       0,1,2,3,4,5,6,7,8,9,10,11,12,13,14,15);
        __builtin_amdgcn_wave_barrier();

        // O += P(16x32) * V(32x64), V pre-transposed => contiguous B-fragments
        int vstart = kv0 + hl * 16;
        if (vstart > L_ - 16) vstart = L_ - 16;  // masked region; addr safety
        const bf16* vbase = vt + (size_t)((b * NH_ + h) * DH_) * L_ + vstart;
        for (int c = 0; c < 4; ++c) {
            const bf16* vp = vbase + (size_t)(c * 16 + l16) * L_;
            v16bf vb = *reinterpret_cast<const v16bf*>(vp);
            o[c] = wmma_bf16(pa, vb, o[c]);
        }
    }

    for (int r = 0; r < 8; ++r) {
        const float inv = 1.0f / l_r[r];
        const size_t row = (size_t)(b * L_ + q0 + hl * 8 + r);
        for (int c = 0; c < 4; ++c)
            attnout[row * (NH_ * DH_) + h * DH_ + c * 16 + l16] = f2bf(o[c][r] * inv);
    }
}

// ---------------- host orchestration ----------------
extern "C" void kernel_launch(void* const* d_in, const int* in_sizes, int n_in,
                              void* d_out, int out_size, void* d_ws, size_t ws_size,
                              hipStream_t stream) {
    (void)in_sizes; (void)n_in; (void)out_size; (void)ws_size;
    const float* x    = (const float*)d_in[0];
    const float* Wa   = (const float*)d_in[1];
    const float* Wout = (const float*)d_in[2];
    const float* bout = (const float*)d_in[3];
    const float* W1   = (const float*)d_in[4];
    const float* b1   = (const float*)d_in[5];
    const float* W2   = (const float*)d_in[6];
    const float* b2   = (const float*)d_in[7];
    const float* g1   = (const float*)d_in[8];
    const float* be1  = (const float*)d_in[9];
    const float* g2   = (const float*)d_in[10];
    const float* be2  = (const float*)d_in[11];
    float* out = (float*)d_out;

    char* ws = (char*)d_ws;
    size_t off = 0;
    auto alloc = [&](size_t bytes) -> char* {
        char* p = ws + off;
        off += (bytes + 255) & ~(size_t)255;
        return p;
    };

    // persistent bf16 transposed weights
    bf16* WaT   = (bf16*)alloc((size_t)E_ * NH_ * 3 * DH_ * 2);   // 3072x1024
    bf16* WoutT = (bf16*)alloc((size_t)E_ * E_ * 2);              // 1024x1024
    bf16* W1T   = (bf16*)alloc((size_t)E_ * HID_ * 2);            // 4096x1024
    bf16* W2T   = (bf16*)alloc((size_t)HID_ * E_ * 2);            // 1024x4096
    // activations
    bf16*  lnbuf   = (bf16*)alloc((size_t)M_ * E_ * 2);           // ln1, later ln2
    float* y1      = (float*)alloc((size_t)M_ * E_ * 4);          // x + attn
    bf16*  attnout = (bf16*)alloc((size_t)M_ * E_ * 2);
    // big region: qkv + vt, later reused as FFN hidden
    size_t qkvBytes = (size_t)M_ * (NH_ * 3 * DH_) * 2;           // 24 MiB
    size_t vtBytes  = (size_t)B_ * NH_ * DH_ * L_ * 2;            //  8 MiB
    size_t hBytes   = (size_t)M_ * HID_ * 2;                      // 32 MiB
    size_t bigBytes = qkvBytes + 256 + vtBytes;
    if (hBytes > bigBytes) bigBytes = hBytes;
    char* big  = alloc(bigBytes);
    bf16* qkv  = (bf16*)big;
    bf16* vt   = (bf16*)(big + ((qkvBytes + 255) & ~(size_t)255));
    bf16* hbuf = (bf16*)big;                                      // reuse after attention

    // 1) weight convert + transpose
    wtrans_kernel<<<(E_ * NH_ * 3 * DH_) / 256, 256, 0, stream>>>(Wa,   WaT,   E_,   NH_ * 3 * DH_);
    wtrans_kernel<<<(E_ * E_)            / 256, 256, 0, stream>>>(Wout, WoutT, E_,   E_);
    wtrans_kernel<<<(E_ * HID_)          / 256, 256, 0, stream>>>(W1,   W1T,   E_,   HID_);
    wtrans_kernel<<<(HID_ * E_)          / 256, 256, 0, stream>>>(W2,   W2T,   HID_, E_);

    // 2) LN1
    ln_kernel<<<M_, 256, 0, stream>>>(x, g1, be1, lnbuf);

    // 3) QKV projection (no bias)
    gemm_wmma_kernel<0><<<dim3((NH_ * 3 * DH_) / 128, M_ / 64), 128, 0, stream>>>(
        lnbuf, WaT, nullptr, nullptr, qkv, nullptr, M_, NH_ * 3 * DH_, E_);

    // 4) transpose V for contiguous B-fragments
    build_vt_kernel<<<(B_ * NH_ * DH_ * L_) / 256, 256, 0, stream>>>(qkv, vt);

    // 5) causal flash attention
    attn_kernel<<<(B_ * NH_ * (L_ / 16)) / 4, 128, 0, stream>>>(qkv, vt, attnout);

    // 6) output projection + bout + residual x -> y1 (f32)
    gemm_wmma_kernel<2><<<dim3(E_ / 128, M_ / 64), 128, 0, stream>>>(
        attnout, WoutT, bout, x, nullptr, y1, M_, E_, E_);

    // 7) LN2
    ln_kernel<<<M_, 256, 0, stream>>>(y1, g2, be2, lnbuf);

    // 8) FFN up (identity activation) + b1
    gemm_wmma_kernel<1><<<dim3(HID_ / 128, M_ / 64), 128, 0, stream>>>(
        lnbuf, W1T, b1, nullptr, hbuf, nullptr, M_, HID_, E_);

    // 9) FFN down + b2 + residual x -> out (f32)
    gemm_wmma_kernel<2><<<dim3(E_ / 128, M_ / 64), 128, 0, stream>>>(
        hbuf, W2T, b2, x, nullptr, out, M_, E_, HID_);
}
